// PairformerStack_81862076661849
// MI455X (gfx1250) — matrix operand order repaired and checked by reading, not compiled, not run on hardware
//
#include <hip/hip_runtime.h>

// ---------------------------------------------------------------------------
// Types for CDNA5 WMMA (wave32, v_wmma_f32_16x16x32_bf16)
// ---------------------------------------------------------------------------
typedef __attribute__((ext_vector_type(16))) __bf16 v16bf;
typedef __attribute__((ext_vector_type(8)))  __bf16 v8bf;
typedef __attribute__((ext_vector_type(8)))  float  v8f;

__device__ __forceinline__ v8f wmma_bf16(v16bf a, v16bf b, v8f c) {
    return __builtin_amdgcn_wmma_f32_16x16x32_bf16(false, a, false, b, (short)0, c,
                                                   false, false);
}

__device__ __forceinline__ float actf(float x, int act) {
    if (act == 1) return 1.f / (1.f + __expf(-x));          // sigmoid
    if (act == 2) return x / (1.f + __expf(-x));            // silu
    return x;
}

// ---------------------------------------------------------------------------
// Weight transpose + f32->bf16:  dst[Npad][K] = src[K][N] (zero pad n>=N)
// ---------------------------------------------------------------------------
__global__ void wt_transpose_k(const float* __restrict__ src, __bf16* __restrict__ dst,
                               int K, int N, int Npad) {
    long idx = (long)blockIdx.x * 256 + threadIdx.x;
    if (idx >= (long)Npad * K) return;
    int n = (int)(idx / K);
    int k = (int)(idx % K);
    dst[idx] = (n < N) ? (__bf16)src[(long)k * N + n] : (__bf16)0.0f;
}

// ---------------------------------------------------------------------------
// LayerNorm rows:  out_bf16[M][C] = LN(x)[c]*g[c]+b[c].  One wave per row.
// ---------------------------------------------------------------------------
__global__ void ln_rows_k(const float* __restrict__ x, const float* __restrict__ g,
                          const float* __restrict__ b, __bf16* __restrict__ out, int C) {
    long row = (long)blockIdx.x * 8 + (threadIdx.x >> 5);
    int lane = threadIdx.x & 31;
    const float* xr = x + row * C;
    int per = C >> 5;
    float s = 0.f, s2 = 0.f;
    for (int t = 0; t < per; ++t) {
        float v = xr[t * 32 + lane];
        s += v; s2 += v * v;
    }
    for (int m = 16; m >= 1; m >>= 1) {
        s  += __shfl_xor(s,  m, 32);
        s2 += __shfl_xor(s2, m, 32);
    }
    float mu = s / C;
    float var = s2 / C - mu * mu;
    float r = rsqrtf(var + 1e-5f);
    __bf16* orow = out + row * C;
    for (int t = 0; t < per; ++t) {
        int c = t * 32 + lane;
        orow[c] = (__bf16)((xr[c] - mu) * r * g[c] + b[c]);
    }
}

// ---------------------------------------------------------------------------
// GEMM: out[M][N] (+bias[N]) (+addsrc[M][N]) = act(A[M][K] @ Wt[N][K]^T)
// A, Wt bf16; f32 accumulate via WMMA. Each wave computes a 16 x (16*NT)
// register tile so the A fragment is reused NT times per k-step.
// Optional batch (grid.y) with element strides sA/sB/sO.  resid: out += acc.
// ---------------------------------------------------------------------------
template <typename OutT, int NT>
__global__ void gemm_bf16_kernel(const __bf16* __restrict__ A, const __bf16* __restrict__ Wt,
                                 OutT* __restrict__ out, const float* __restrict__ bias,
                                 const float* __restrict__ addsrc,
                                 int M, int N, int K, int act, int resid,
                                 long sA, long sB, long sO) {
    int ntw = N / (16 * NT);
    int tiles = (M >> 4) * ntw;
    int wt = blockIdx.x * 4 + (threadIdx.x >> 5);
    if (wt >= tiles) return;
    int tm = wt / ntw, tn0 = (wt % ntw) * NT;
    long bb = blockIdx.y;
    A += bb * sA; Wt += bb * sB; out += bb * sO;
    if (addsrc) addsrc += bb * sO;
    int lane = threadIdx.x & 31, half = lane >> 4, l16 = lane & 15;
    const __bf16* arow = A + (long)(tm * 16 + l16) * K + half * 16;
    const __bf16* brow[NT];
#pragma unroll
    for (int j = 0; j < NT; ++j)
        brow[j] = Wt + (long)((tn0 + j) * 16 + l16) * K + half * 16;
    v8f acc[NT];
#pragma unroll
    for (int j = 0; j < NT; ++j) acc[j] = (v8f){0, 0, 0, 0, 0, 0, 0, 0};
    for (int k0 = 0; k0 < K; k0 += 32) {
        __builtin_prefetch(arow + k0 + 256, 0, 3);
        v8bf alo = *(const v8bf*)(arow + k0);
        v8bf ahi = *(const v8bf*)(arow + k0 + 8);
        v16bf af = __builtin_shufflevector(alo, ahi, 0, 1, 2, 3, 4, 5, 6, 7,
                                           8, 9, 10, 11, 12, 13, 14, 15);
#pragma unroll
        for (int j = 0; j < NT; ++j)
            acc[j] = wmma_bf16(af, *(const v16bf*)(brow[j] + k0), acc[j]);
    }
    int gm0 = tm * 16 + half * 8;
#pragma unroll
    for (int j = 0; j < NT; ++j) {
        int gn = (tn0 + j) * 16 + l16;
#pragma unroll
        for (int r = 0; r < 8; ++r) {
            long idx = (long)(gm0 + r) * N + gn;
            float v = acc[j][r];
            if (bias)   v += bias[gn];
            if (addsrc) v += addsrc[idx];
            v = actf(v, act);
            if (resid)  v += (float)out[idx];
            out[idx] = (OutT)v;
        }
    }
}

// Gated GEMM: out_bf16 = act(A@W1t^T) * (A@W2t^T). 2 N-tiles x 2 weights
// per wave -> 4 WMMA per A-fragment load.  Requires N % 32 == 0.
__global__ void gemm_gated_kernel(const __bf16* __restrict__ A, const __bf16* __restrict__ W1,
                                  const __bf16* __restrict__ W2, __bf16* __restrict__ out,
                                  int M, int N, int K, int act) {
    int ntw = N >> 5;                      // 32 columns per wave
    int tiles = (M >> 4) * ntw;
    int wt = blockIdx.x * 4 + (threadIdx.x >> 5);
    if (wt >= tiles) return;
    int tm = wt / ntw, tn0 = (wt % ntw) * 2;
    int lane = threadIdx.x & 31, half = lane >> 4, l16 = lane & 15;
    const __bf16* arow = A + (long)(tm * 16 + l16) * K + half * 16;
    const __bf16* b1[2];
    const __bf16* b2[2];
#pragma unroll
    for (int j = 0; j < 2; ++j) {
        long roff = (long)((tn0 + j) * 16 + l16) * K + half * 16;
        b1[j] = W1 + roff;
        b2[j] = W2 + roff;
    }
    v8f acc1[2] = {{0, 0, 0, 0, 0, 0, 0, 0}, {0, 0, 0, 0, 0, 0, 0, 0}};
    v8f acc2[2] = {{0, 0, 0, 0, 0, 0, 0, 0}, {0, 0, 0, 0, 0, 0, 0, 0}};
    for (int k0 = 0; k0 < K; k0 += 32) {
        __builtin_prefetch(arow + k0 + 256, 0, 3);
        v8bf alo = *(const v8bf*)(arow + k0);
        v8bf ahi = *(const v8bf*)(arow + k0 + 8);
        v16bf af = __builtin_shufflevector(alo, ahi, 0, 1, 2, 3, 4, 5, 6, 7,
                                           8, 9, 10, 11, 12, 13, 14, 15);
#pragma unroll
        for (int j = 0; j < 2; ++j) {
            acc1[j] = wmma_bf16(af, *(const v16bf*)(b1[j] + k0), acc1[j]);
            acc2[j] = wmma_bf16(af, *(const v16bf*)(b2[j] + k0), acc2[j]);
        }
    }
    int gm0 = tm * 16 + half * 8;
#pragma unroll
    for (int j = 0; j < 2; ++j) {
        int gn = (tn0 + j) * 16 + l16;
#pragma unroll
        for (int r = 0; r < 8; ++r) {
            long idx = (long)(gm0 + r) * N + gn;
            out[idx] = (__bf16)(actf(acc1[j][r], act) * acc2[j][r]);
        }
    }
}

// ---------------------------------------------------------------------------
// Triangle einsum data movement
// dst[c][x][y] = src[(swap ? y*192+x : x*192+y)*128 + c]     (bf16 -> bf16)
// ---------------------------------------------------------------------------
__global__ void tri_tr_ab_k(const __bf16* __restrict__ src, __bf16* __restrict__ dst, int swp) {
    long idx = (long)blockIdx.x * 256 + threadIdx.x;
    if (idx >= 128L * 36864) return;
    int y = (int)(idx % 192);
    long t = idx / 192;
    int x = (int)(t % 192);
    int c = (int)(t / 192);
    long sidx = swp ? ((long)(y * 192 + x) * 128 + c) : ((long)(x * 192 + y) * 128 + c);
    dst[idx] = src[sidx];
}

// dst[(ij)*128 + c] = src[c*36864 + ij]   (f32)
__global__ void tri_tr_back_k(const float* __restrict__ src, float* __restrict__ dst) {
    long idx = (long)blockIdx.x * 256 + threadIdx.x;
    if (idx >= 36864L * 128) return;
    int c = (int)(idx % 128);
    long ij = idx / 128;
    dst[idx] = src[(long)c * 36864 + ij];
}

// vt[((outer*4+h)*32+d)*192 + k] = vraw[gidx(k)*128 + h*32 + d]
__global__ void v_transpose_k(const __bf16* __restrict__ vraw, __bf16* __restrict__ vt,
                              int ending) {
    long idx = (long)blockIdx.x * 256 + threadIdx.x;
    if (idx >= 36864L * 128) return;
    int k = (int)(idx % 192);
    long t = idx / 192;
    int d = (int)(t & 31); t >>= 5;
    int h = (int)(t & 3);
    int outer = (int)(t >> 2);
    long g = ending ? ((long)k * 192 + outer) : ((long)outer * 192 + k);
    vt[idx] = vraw[g * 128 + h * 32 + d];
}

// ---------------------------------------------------------------------------
// Fused triangle attention: one wave = (outer, head, q-tile of 16 rows),
// all 192 keys. WMMA scores -> in-register softmax -> WMMA P@V via LDS stage.
// ---------------------------------------------------------------------------
__global__ void tri_attn_k(const __bf16* __restrict__ qb, const __bf16* __restrict__ kb,
                           const __bf16* __restrict__ vt, const float* __restrict__ bias,
                           float* __restrict__ ob, int ending) {
    int id = blockIdx.x;
    int qt = id % 12;
    int h = (id / 12) & 3;
    int outer = id / 48;
    int lane = threadIdx.x, half = lane >> 4, l16 = lane & 15;
    const float inv = 0.17677669529663687f;  // 1/sqrt(32)

    int qrowl = qt * 16 + l16;
    long qg = ending ? ((long)qrowl * 192 + outer) : ((long)outer * 192 + qrowl);
    const __bf16* qp = qb + qg * 128 + h * 32 + half * 16;
    v8bf qlo = *(const v8bf*)qp;
    v8bf qhi = *(const v8bf*)(qp + 8);
    v16bf qf = __builtin_shufflevector(qlo, qhi, 0, 1, 2, 3, 4, 5, 6, 7,
                                       8, 9, 10, 11, 12, 13, 14, 15);
    float sc[12][8];
    for (int t = 0; t < 12; ++t) {
        int key = t * 16 + l16;
        long kg = ending ? ((long)key * 192 + outer) : ((long)outer * 192 + key);
        v16bf kf = *(const v16bf*)(kb + kg * 128 + h * 32 + half * 16);
        v8f z = {0, 0, 0, 0, 0, 0, 0, 0};
        v8f s = wmma_bf16(qf, kf, z);
        for (int r = 0; r < 8; ++r) {
            int qm = qt * 16 + half * 8 + r;   // query row (global within 192)
            int kgl = t * 16 + l16;            // key index (this lane's score col)
            long bidx = ending ? ((long)kgl * 192 + qm) : ((long)qm * 192 + kgl);
            sc[t][r] = s[r] * inv + bias[bidx * 16 + h];
        }
    }
    // softmax over all 192 keys per query row (rows live in 16-lane halves)
    for (int r = 0; r < 8; ++r) {
        float mx = -3.4e38f;
        for (int t = 0; t < 12; ++t) mx = fmaxf(mx, sc[t][r]);
        for (int m = 8; m >= 1; m >>= 1) mx = fmaxf(mx, __shfl_xor(mx, m, 32));
        float sum = 0.f;
        for (int t = 0; t < 12; ++t) {
            float e = __expf(sc[t][r] - mx);
            sc[t][r] = e; sum += e;
        }
        for (int m = 8; m >= 1; m >>= 1) sum += __shfl_xor(sum, m, 32);
        float rs = 1.f / sum;
        for (int t = 0; t < 12; ++t) sc[t][r] *= rs;
    }
    // P @ V: stage 16x32 prob tiles through LDS to get A-fragment layout
    __shared__ __bf16 lds[16 * 32];
    v8f o0 = {0, 0, 0, 0, 0, 0, 0, 0};
    v8f o1 = {0, 0, 0, 0, 0, 0, 0, 0};
    long vbase = ((long)outer * 4 + h) * 32;
    for (int kt = 0; kt < 6; ++kt) {
        for (int tt = 0; tt < 2; ++tt) {
            int t = kt * 2 + tt;
            for (int r = 0; r < 8; ++r)
                lds[(half * 8 + r) * 32 + tt * 16 + l16] = (__bf16)sc[t][r];
        }
        __syncthreads();
        const __bf16* lp = &lds[l16 * 32 + half * 16];
        v8bf alo = *(const v8bf*)lp;
        v8bf ahi = *(const v8bf*)(lp + 8);
        v16bf af = __builtin_shufflevector(alo, ahi, 0, 1, 2, 3, 4, 5, 6, 7,
                                           8, 9, 10, 11, 12, 13, 14, 15);
        v16bf vf0 = *(const v16bf*)(vt + (vbase + l16) * 192 + kt * 32 + half * 16);
        v16bf vf1 = *(const v16bf*)(vt + (vbase + 16 + l16) * 192 + kt * 32 + half * 16);
        o0 = wmma_bf16(af, vf0, o0);
        o1 = wmma_bf16(af, vf1, o1);
        __syncthreads();
    }
    for (int r = 0; r < 8; ++r) {
        int qm = qt * 16 + half * 8 + r;
        long g = ending ? ((long)qm * 192 + outer) : ((long)outer * 192 + qm);
        long base = g * 128 + h * 32 + l16;
        ob[base] = o0[r];
        ob[base + 16] = o1[r];
    }
}

// ---------------------------------------------------------------------------
// Pair-bias attention for s (DA=24, HA=16): one wave per (i, h). VALU math.
// ---------------------------------------------------------------------------
__global__ void apb_attn_k(const float* __restrict__ qa, const float* __restrict__ ka,
                           const float* __restrict__ va, const float* __restrict__ ga,
                           const float* __restrict__ bz, __bf16* __restrict__ satt) {
    int w = blockIdx.x * 8 + (threadIdx.x >> 5);
    int lane = threadIdx.x & 31;
    int h = w & 15, i = w >> 4;
    __shared__ float probs[8][192];
    float* P = probs[threadIdx.x >> 5];
    const float inv = 0.20412414523193154f;  // 1/sqrt(24)
    float qv[24];
    for (int d = 0; d < 24; ++d) qv[d] = qa[i * 384 + h * 24 + d];
    float sc[6];
    for (int it = 0; it < 6; ++it) {
        int j = it * 32 + lane;
        float s = 0.f;
        for (int d = 0; d < 24; ++d) s += qv[d] * ka[j * 384 + h * 24 + d];
        sc[it] = s * inv + bz[((long)i * 192 + j) * 16 + h];
    }
    float mx = -3.4e38f;
    for (int it = 0; it < 6; ++it) mx = fmaxf(mx, sc[it]);
    for (int m = 16; m >= 1; m >>= 1) mx = fmaxf(mx, __shfl_xor(mx, m, 32));
    float sum = 0.f;
    for (int it = 0; it < 6; ++it) { sc[it] = __expf(sc[it] - mx); sum += sc[it]; }
    for (int m = 16; m >= 1; m >>= 1) sum += __shfl_xor(sum, m, 32);
    float rs = 1.f / sum;
    for (int it = 0; it < 6; ++it) P[it * 32 + lane] = sc[it] * rs;
    __syncthreads();
    if (lane < 24) {
        float o = 0.f;
        for (int j = 0; j < 192; ++j) o += P[j] * va[j * 384 + h * 24 + lane];
        o *= ga[i * 384 + h * 24 + lane];
        satt[i * 384 + h * 24 + lane] = (__bf16)o;
    }
}

// ---------------------------------------------------------------------------
// Elementwise helpers
// ---------------------------------------------------------------------------
__global__ void mul_to_bf16_k(const float* __restrict__ a, const float* __restrict__ b,
                              __bf16* __restrict__ o, long n) {
    long i = (long)blockIdx.x * 256 + threadIdx.x;
    if (i < n) o[i] = (__bf16)(a[i] * b[i]);
}
__global__ void fma_inplace_k(float* __restrict__ dst, const float* __restrict__ a,
                              const float* __restrict__ b, long n) {
    long i = (long)blockIdx.x * 256 + threadIdx.x;
    if (i < n) dst[i] += a[i] * b[i];
}

// ---------------------------------------------------------------------------
// Host orchestration
// ---------------------------------------------------------------------------
extern "C" void kernel_launch(void* const* d_in, const int* in_sizes, int n_in,
                              void* d_out, int out_size, void* d_ws, size_t ws_size,
                              hipStream_t stream) {
    (void)in_sizes; (void)n_in; (void)out_size; (void)ws_size;
    const long Mz = 36864;  // 192*192

    // ---- inputs (top level in setup_inputs order; params jax-tree sorted) ----
    const float* s_in = (const float*)d_in[0];
    const float* z_in = (const float*)d_in[1];
    const float* beta = (const float*)d_in[2];
    int p = 3;
    auto nxt = [&]() { return (const float*)d_in[p++]; };
    // apb: bw, gw, kw, lna_b, lna_g, lnb_b, lnb_g, ow, qb, qw, vw
    const float *A_bw = nxt(), *A_gw = nxt(), *A_kw = nxt(), *A_lnab = nxt(), *A_lnag = nxt();
    const float *A_lnbb = nxt(), *A_lnbg = nxt(), *A_ow = nxt(), *A_qb = nxt(), *A_qw = nxt(), *A_vw = nxt();
    // tae: b, g, k, ln_b, ln_g, out, q, v
    const float *TAE_b = nxt(), *TAE_g = nxt(), *TAE_k = nxt(), *TAE_lnb = nxt(), *TAE_lng = nxt(),
                *TAE_out = nxt(), *TAE_q = nxt(), *TAE_v = nxt();
    // tas
    const float *TAS_b = nxt(), *TAS_g = nxt(), *TAS_k = nxt(), *TAS_lnb = nxt(), *TAS_lng = nxt(),
                *TAS_out = nxt(), *TAS_q = nxt(), *TAS_v = nxt();
    // tmi: a1, a2, b1, b2, g, ln_b, ln_g, ob, og, out
    const float *TMI_a1 = nxt(), *TMI_a2 = nxt(), *TMI_b1 = nxt(), *TMI_b2 = nxt(), *TMI_g = nxt(),
                *TMI_lnb = nxt(), *TMI_lng = nxt(), *TMI_ob = nxt(), *TMI_og = nxt(), *TMI_out = nxt();
    // tmo
    const float *TMO_a1 = nxt(), *TMO_a2 = nxt(), *TMO_b1 = nxt(), *TMO_b2 = nxt(), *TMO_g = nxt(),
                *TMO_lnb = nxt(), *TMO_lng = nxt(), *TMO_ob = nxt(), *TMO_og = nxt(), *TMO_out = nxt();
    // trs: a, b, ln_b, ln_g, out
    const float *TRS_a = nxt(), *TRS_b = nxt(), *TRS_lnb = nxt(), *TRS_lng = nxt(), *TRS_out = nxt();
    // trz
    const float *TRZ_a = nxt(), *TRZ_b = nxt(), *TRZ_lnb = nxt(), *TRZ_lng = nxt(), *TRZ_out = nxt();

    // ---- live state lives in d_out: [s (192*384) | z (36864*128)] ----
    float* S = (float*)d_out;
    float* Z = (float*)d_out + 192 * 384;
    hipMemcpyAsync(S, s_in, 192 * 384 * sizeof(float), hipMemcpyDeviceToDevice, stream);
    hipMemcpyAsync(Z, z_in, Mz * 128 * sizeof(float), hipMemcpyDeviceToDevice, stream);

    // ---- workspace carving ----
    char* wsb = (char*)d_ws;
    auto carve = [&](size_t bytes) { char* q = wsb; wsb += (bytes + 255) & ~(size_t)255; return q; };
    const size_t WSLOT = 1536 * 384;  // largest Wt (elements)
    __bf16* w0 = (__bf16*)carve(WSLOT * 2); __bf16* w1 = (__bf16*)carve(WSLOT * 2);
    __bf16* w2 = (__bf16*)carve(WSLOT * 2); __bf16* w3 = (__bf16*)carve(WSLOT * 2);
    __bf16* w4 = (__bf16*)carve(WSLOT * 2); __bf16* w5 = (__bf16*)carve(WSLOT * 2);
    __bf16* ZN  = (__bf16*)carve(Mz * 128 * 2);
    __bf16* BF1 = (__bf16*)carve(Mz * 128 * 2);
    __bf16* BF2 = (__bf16*)carve(Mz * 128 * 2);
    __bf16* BF3 = (__bf16*)carve(Mz * 128 * 2);
    __bf16* BF4 = (__bf16*)carve(Mz * 128 * 2);
    __bf16* BBIG = (__bf16*)carve(Mz * 512 * 2);  // transition gated / vraw+vt
    float* F1 = (float*)carve(Mz * 128 * 4);
    float* F2 = (float*)carve(Mz * 128 * 4);
    float* F3 = (float*)carve(Mz * 128 * 4);
    float* F4 = (float*)carve(Mz * 16 * 4);
    __bf16* SB1  = (__bf16*)carve(192 * 1536 * 2);
    __bf16* SB2  = (__bf16*)carve(192 * 1536 * 2);
    float* SQ = (float*)carve(192 * 384 * 4);
    float* SK = (float*)carve(192 * 384 * 4);
    float* SV = (float*)carve(192 * 384 * 4);
    float* SG = (float*)carve(192 * 384 * 4);
    __bf16* SATT = (__bf16*)carve(192 * 384 * 2);
    __bf16* VRAW = BBIG;
    __bf16* VT   = BBIG + Mz * 128;

    // ---- launch helpers ----
    auto LN = [&](const float* x, const float* g, const float* b, __bf16* o, long M, int C) {
        ln_rows_k<<<dim3((unsigned)(M / 8)), 256, 0, stream>>>(x, g, b, o, C);
    };
    auto WT = [&](const float* w, int l, int K, int N, int Npad, __bf16* dst) {
        long tot = (long)Npad * K;
        wt_transpose_k<<<dim3((unsigned)((tot + 255) / 256)), 256, 0, stream>>>(
            w + (long)l * K * N, dst, K, N, Npad);
    };
    auto GEMMF = [&](const __bf16* A, const __bf16* W, float* out, const float* bias,
                     const float* addsrc, long M, int N, int K, int act, int resid,
                     int batch, long sA, long sB, long sO) {
        if (N % 64 == 0) {
            long tiles = (M / 16) * (N / 64);
            dim3 g((unsigned)((tiles + 3) / 4), (unsigned)batch);
            gemm_bf16_kernel<float, 4><<<g, 128, 0, stream>>>(
                A, W, out, bias, addsrc, (int)M, N, K, act, resid, sA, sB, sO);
        } else {
            long tiles = (M / 16) * (N / 16);
            dim3 g((unsigned)((tiles + 3) / 4), (unsigned)batch);
            gemm_bf16_kernel<float, 1><<<g, 128, 0, stream>>>(
                A, W, out, bias, addsrc, (int)M, N, K, act, resid, sA, sB, sO);
        }
    };
    auto GEMMB = [&](const __bf16* A, const __bf16* W, __bf16* out, long M, int N, int K) {
        long tiles = (M / 16) * (N / 64);
        gemm_bf16_kernel<__bf16, 4><<<dim3((unsigned)((tiles + 3) / 4)), 128, 0, stream>>>(
            A, W, out, nullptr, nullptr, (int)M, N, K, 0, 0, 0, 0, 0);
    };
    auto GGATE = [&](const __bf16* A, const __bf16* Wa, const __bf16* Wb, __bf16* out,
                     long M, int N, int K, int act) {
        long tiles = (M / 16) * (N / 32);
        gemm_gated_kernel<<<dim3((unsigned)((tiles + 3) / 4)), 128, 0, stream>>>(
            A, Wa, Wb, out, (int)M, N, K, act);
    };
    const unsigned EWG = (unsigned)((Mz * 128 + 255) / 256);

    struct TMp { const float *a1, *a2, *b1, *b2, *g, *lnb, *lng, *ob, *og, *out; int swp; };
    struct TAp { const float *b, *g, *k, *lnb, *lng, *out, *q, *v; int ending; };

    for (int l = 0; l < 2; ++l) {
        // ================= triangle multiplications (outgoing, incoming) ======
        TMp TM[2] = {
            {TMO_a1, TMO_a2, TMO_b1, TMO_b2, TMO_g, TMO_lnb, TMO_lng, TMO_ob, TMO_og, TMO_out, 0},
            {TMI_a1, TMI_a2, TMI_b1, TMI_b2, TMI_g, TMI_lnb, TMI_lng, TMI_ob, TMI_og, TMI_out, 1}};
        for (int d = 0; d < 2; ++d) {
            const TMp& t = TM[d];
            LN(Z, t.lng + l * 128, t.lnb + l * 128, ZN, Mz, 128);
            WT(t.a1, l, 128, 128, 128, w0); WT(t.a2, l, 128, 128, 128, w1);
            WT(t.b1, l, 128, 128, 128, w2); WT(t.b2, l, 128, 128, 128, w3);
            WT(t.g,  l, 128, 128, 128, w4); WT(t.out, l, 128, 128, 128, w5);
            GGATE(ZN, w0, w1, BF1, Mz, 128, 128, 1);              // a = sig(.)*(.)
            GGATE(ZN, w2, w3, BF2, Mz, 128, 128, 1);              // b
            tri_tr_ab_k<<<EWG, 256, 0, stream>>>(BF1, BF3, t.swp);
            tri_tr_ab_k<<<EWG, 256, 0, stream>>>(BF2, BF4, t.swp);
            GEMMF(BF3, BF4, F2, nullptr, nullptr, 192, 192, 192, 0, 0,
                  128, 36864, 36864, 36864);                       // per-channel einsum
            tri_tr_back_k<<<EWG, 256, 0, stream>>>(F2, F3);        // sab [Mz,128]
            LN(F3, t.og + l * 128, t.ob + l * 128, BF1, Mz, 128);
            GEMMF(BF1, w5, F2, nullptr, nullptr, Mz, 128, 128, 0, 0, 1, 0, 0, 0);  // tmp
            GEMMF(ZN, w4, F1, nullptr, nullptr, Mz, 128, 128, 1, 0, 1, 0, 0, 0);   // gate
            fma_inplace_k<<<EWG, 256, 0, stream>>>(Z, F1, F2, Mz * 128);           // z += g*tmp
        }
        // ================= triangle attentions (starting, ending) =============
        TAp TA[2] = {
            {TAS_b, TAS_g, TAS_k, TAS_lnb, TAS_lng, TAS_out, TAS_q, TAS_v, 0},
            {TAE_b, TAE_g, TAE_k, TAE_lnb, TAE_lng, TAE_out, TAE_q, TAE_v, 1}};
        for (int d = 0; d < 2; ++d) {
            const TAp& t = TA[d];
            LN(Z, t.lng + l * 128, t.lnb + l * 128, ZN, Mz, 128);
            WT(t.q, l, 128, 128, 128, w0); WT(t.k, l, 128, 128, 128, w1);
            WT(t.v, l, 128, 128, 128, w2); WT(t.g, l, 128, 128, 128, w3);
            WT(t.out, l, 128, 128, 128, w4); WT(t.b, l, 128, 4, 16, w5);
            GEMMB(ZN, w0, BF1, Mz, 128, 128);                         // q (bf16)
            GEMMB(ZN, w1, BF2, Mz, 128, 128);                         // k
            GEMMB(ZN, w2, VRAW, Mz, 128, 128);                        // v
            GEMMF(ZN, w3, F1, nullptr, nullptr, Mz, 128, 128, 1, 0, 1, 0, 0, 0);  // gate (sig)
            GEMMF(ZN, w5, F4, nullptr, nullptr, Mz, 16, 128, 0, 0, 1, 0, 0, 0);   // bias (pad16)
            v_transpose_k<<<EWG, 256, 0, stream>>>(VRAW, VT, t.ending);
            tri_attn_k<<<dim3(192 * 4 * 12), 32, 0, stream>>>(BF1, BF2, VT, F4, F2, t.ending);
            mul_to_bf16_k<<<EWG, 256, 0, stream>>>(F1, F2, BF3, Mz * 128);  // g*o
            GEMMF(BF3, w4, Z, nullptr, nullptr, Mz, 128, 128, 0, 1, 1, 0, 0, 0);  // z += .@out
        }
        // ================= pair transition ====================================
        LN(Z, TRZ_lng + l * 128, TRZ_lnb + l * 128, ZN, Mz, 128);
        WT(TRZ_a, l, 128, 512, 512, w0);
        WT(TRZ_b, l, 128, 512, 512, w1);
        WT(TRZ_out, l, 512, 128, 128, w2);
        GGATE(ZN, w0, w1, BBIG, Mz, 512, 128, 2);                                  // silu gate
        GEMMF(BBIG, w2, Z, nullptr, nullptr, Mz, 128, 512, 0, 1, 1, 0, 0, 0);      // z += .
        // ================= attention-pair-bias on s ===========================
        LN(S, A_lnag + l * 384, A_lnab + l * 384, SB1, 192, 384);
        WT(A_qw, l, 384, 384, 384, w0); WT(A_kw, l, 384, 384, 384, w1);
        WT(A_vw, l, 384, 384, 384, w2); WT(A_gw, l, 384, 384, 384, w3);
        WT(A_ow, l, 384, 384, 384, w4); WT(A_bw, l, 128, 16, 16, w5);
        GEMMF(SB1, w0, SQ, A_qb + l * 384, nullptr, 192, 384, 384, 0, 0, 1, 0, 0, 0);
        GEMMF(SB1, w1, SK, nullptr, nullptr, 192, 384, 384, 0, 0, 1, 0, 0, 0);
        GEMMF(SB1, w2, SV, nullptr, nullptr, 192, 384, 384, 0, 0, 1, 0, 0, 0);
        GEMMF(SB1, w3, SG, nullptr, nullptr, 192, 384, 384, 1, 0, 1, 0, 0, 0);
        LN(Z, A_lnbg + l * 128, A_lnbb + l * 128, ZN, Mz, 128);
        GEMMF(ZN, w5, F4, nullptr, beta, Mz, 16, 128, 0, 0, 1, 0, 0, 0);   // bz + beta
        apb_attn_k<<<dim3(384), 256, 0, stream>>>(SQ, SK, SV, SG, F4, SATT);
        GEMMF(SATT, w4, S, nullptr, nullptr, 192, 384, 384, 0, 1, 1, 0, 0, 0);  // s += .
        // ================= single transition ==================================
        LN(S, TRS_lng + l * 384, TRS_lnb + l * 384, SB1, 192, 384);
        WT(TRS_a, l, 384, 1536, 1536, w0);
        WT(TRS_b, l, 384, 1536, 1536, w1);
        WT(TRS_out, l, 1536, 384, 384, w2);
        GGATE(SB1, w0, w1, SB2, 192, 1536, 384, 2);
        GEMMF(SB2, w2, S, nullptr, nullptr, 192, 384, 1536, 0, 1, 1, 0, 0, 0);
    }
}
